// DiffusionDecoder_58454504899320
// MI455X (gfx1250) — compile-verified, matrix operand order checked by reading, not run on hardware
//
#include <hip/hip_runtime.h>
#include <hip/hip_bf16.h>
#include <math.h>

// ---- model dims (match reference) ----
#define Dm   512
#define Hn   8
#define Ln   8
#define En   4
#define PCn  5
#define Bn   8
#define Sn   512
#define CTXn 128
#define DFn  2048
#define HDn  64
#define NTOK (Bn * Sn)   // 4096 tokens

typedef _Float16 f16;
typedef _Float16 v16h __attribute__((ext_vector_type(16)));
typedef float    v8f  __attribute__((ext_vector_type(8)));
typedef float    v4fv __attribute__((ext_vector_type(4)));

// ---- WMMA GEMM tiling ----
#define BM 64
#define BN 32
#define BK 64    // two 16x16x32 WMMAs per staged tile
#define LDT 68   // LDS tile stride in floats (272B: 16B-aligned chunks, bank-skewed)

// C[M,N] = epilogue(act(alpha * A[M,K] @ B[K,N] + bias))
// ACT: 0 none, 1 relu, 2 exact gelu
// EPI: 0 store, 1 store + res, 2 C += rowscale[m] * x   (MoE combine)
// TRANSB: B_logical[k][n] = Bm[n*ldb + k]   (for Q @ K^T)
// Batched via blockIdx.z: zb = z / Hdim, zh = z % Hdim; per-matrix (b,h) strides.
//
// Pipeline: fp32 tiles are DMA'd into double-buffered LDS with the gfx1250
// async-to-LDS path (global_load_async_to_lds_b128/b32). One barrier per
// K-step: wait own ASYNCcnt -> barrier -> issue next tile's asyncs into the
// other buffer -> compute current buffer. Staging pointers advance by BK
// (no per-iteration 64-bit multiplies); buf index is compile-time constant.
// B is staged TRANSPOSED ([n][k]) so A and B fragments are aligned
// ds_load_b128 runs (ISA 7.12.2: lane's 16 halfs = K runs g*8..+7 and
// g*8+16..+23); fp32->f16 converts in-register (co-executes with WMMA).
template <int ACT, int EPI, bool TRANSB, bool HAS_BIAS>
__global__ __launch_bounds__(256) void gemm_wmma(
    const float* __restrict__ A, int lda,
    const float* __restrict__ Bm, int ldb,
    const float* __restrict__ bias,
    const float* __restrict__ res, int ldres,
    const float* __restrict__ rowscale, int rs_stride,
    float* __restrict__ C, int ldc,
    int N, int K, float alpha,
    int Hdim, long sAb, long sAh, long sBb, long sBh, long sCb, long sCh)
{
    __shared__ float Asf[2][BM][LDT];   // A tiles, [m][k]
    __shared__ float Btf[2][BN][LDT];   // B tiles transposed, [n][k]

    const int z  = blockIdx.z;
    const int zb = z / Hdim;
    const int zh = z % Hdim;
    const float* Ap = A  + (long)zb * sAb + (long)zh * sAh;
    const float* Bp = Bm + (long)zb * sBb + (long)zh * sBh;
    float*       Cp = C  + (long)zb * sCb + (long)zh * sCh;

    const int m0   = blockIdx.y * BM;
    const int n0   = blockIdx.x * BN;
    const int tid  = threadIdx.x;
    const int lane = tid & 31;
    const int wave = tid >> 5;    // 0..7
    const int wm   = wave & 3;    // M subtile 0..3
    const int wn   = wave >> 2;   // N subtile 0..1

    v8f acc = {0.f, 0.f, 0.f, 0.f, 0.f, 0.f, 0.f, 0.f};

    const int row = wm * 16 + (lane & 15);   // A fragment row within tile
    const int col = wn * 16 + (lane & 15);   // B fragment col within tile
    const int g8  = (lane >> 4) << 3;        // 0 or 8 (K group per lane-half)

    // staging assignment (fixed per thread): 16B chunks
    const int cm4 = tid >> 4;          // chunk row base (0..15)
    const int ck4 = (tid & 15) << 2;   // chunk k offset (0..60)
    const int bn  = tid & 31;          // b32-transpose lane (non-TRANSB)
    const int bk  = tid >> 5;          // b32-transpose k base (0..7)

    // per-thread global staging pointers; advance by BK each stage
    const float* ga[4];
    #pragma unroll
    for (int i = 0; i < 4; ++i)
        ga[i] = &Ap[(long)(m0 + cm4 + i * 16) * lda + ck4];
    const float* gbt[2];
    const float* gbn[8];
    if (TRANSB) {
        gbt[0] = &Bp[(long)(n0 + cm4) * ldb + ck4];
        gbt[1] = &Bp[(long)(n0 + cm4 + 16) * ldb + ck4];
    } else {
        #pragma unroll
        for (int j = 0; j < 8; ++j)
            gbn[j] = &Bp[(long)(bk + j * 8) * ldb + (n0 + bn)];
    }

    auto stage = [&](int buf) {
        // A tile: 64x64 fp32 = 1024 chunks -> 4 async b128 per thread
        #pragma unroll
        for (int i = 0; i < 4; ++i) {
            unsigned l = (unsigned)(size_t)&Asf[buf][cm4 + i * 16][ck4];
            asm volatile("global_load_async_to_lds_b128 %0, %1, off"
                         :: "v"(l), "v"(ga[i]) : "memory");
            ga[i] += BK;
        }
        if (TRANSB) {
            // global already k-contiguous per n row: 2 async b128 per thread
            #pragma unroll
            for (int i = 0; i < 2; ++i) {
                unsigned l = (unsigned)(size_t)&Btf[buf][cm4 + i * 16][ck4];
                asm volatile("global_load_async_to_lds_b128 %0, %1, off"
                             :: "v"(l), "v"(gbt[i]) : "memory");
                gbt[i] += BK;
            }
        } else {
            // transpose at b32 granularity: 8 async b32 per thread
            #pragma unroll
            for (int j = 0; j < 8; ++j) {
                unsigned l = (unsigned)(size_t)&Btf[buf][bn][bk + j * 8];
                asm volatile("global_load_async_to_lds_b32 %0, %1, off"
                             :: "v"(l), "v"(gbn[j]) : "memory");
                gbn[j] += (long)BK * ldb;
            }
        }
    };

    auto compute_half = [&](int buf, int ku) {
        v4fv a0 = *(const v4fv*)&Asf[buf][row][ku + g8];
        v4fv a1 = *(const v4fv*)&Asf[buf][row][ku + g8 + 4];
        v4fv a2 = *(const v4fv*)&Asf[buf][row][ku + g8 + 16];
        v4fv a3 = *(const v4fv*)&Asf[buf][row][ku + g8 + 20];
        v4fv b0 = *(const v4fv*)&Btf[buf][col][ku + g8];
        v4fv b1 = *(const v4fv*)&Btf[buf][col][ku + g8 + 4];
        v4fv b2 = *(const v4fv*)&Btf[buf][col][ku + g8 + 16];
        v4fv b3 = *(const v4fv*)&Btf[buf][col][ku + g8 + 20];
        v16h af, bf;
        #pragma unroll
        for (int e = 0; e < 4; ++e) {
            af[e]      = (f16)a0[e];
            af[e + 4]  = (f16)a1[e];
            af[e + 8]  = (f16)a2[e];
            af[e + 12] = (f16)a3[e];
            bf[e]      = (f16)b0[e];
            bf[e + 4]  = (f16)b1[e];
            bf[e + 8]  = (f16)b2[e];
            bf[e + 12] = (f16)b3[e];
        }
        acc = __builtin_amdgcn_wmma_f32_16x16x32_f16(false, af, false, bf,
                                                     (short)0, acc, false, false);
    };

    const int nk = K / BK;
    int it = 0;
    stage(0);
    for (;;) {
        // ---- buffer 0 ----
        asm volatile("s_wait_asynccnt 0x0" ::: "memory");
        __syncthreads();
        if (it + 1 < nk) stage(1);
        compute_half(0, 0);
        compute_half(0, 32);
        if (++it == nk) break;
        // ---- buffer 1 ----
        asm volatile("s_wait_asynccnt 0x0" ::: "memory");
        __syncthreads();
        if (it + 1 < nk) stage(0);
        compute_half(1, 0);
        compute_half(1, 32);
        if (++it == nk) break;
    }

    // ---- epilogue: C/D layout VGPR v holds M=v (lanes 0-15) and M=8+v (lanes 16-31) ----
    const int cn = n0 + wn * 16 + (lane & 15);
    const int mb = m0 + wm * 16 + g8;
    const long base = (long)mb * ldc + cn;
    const float bv = HAS_BIAS ? bias[cn] : 0.f;

    float extra[8];
    if (EPI == 1) {
        #pragma unroll
        for (int v = 0; v < 8; ++v) extra[v] = res[(long)(mb + v) * ldres + cn];
    } else if (EPI == 2) {
        #pragma unroll
        for (int v = 0; v < 8; ++v) extra[v] = Cp[base + (long)v * ldc];
    }
    float rs[8];
    if (EPI == 2) {
        #pragma unroll
        for (int v = 0; v < 8; ++v) rs[v] = rowscale[(long)(mb + v) * rs_stride];
    }

    #pragma unroll
    for (int v = 0; v < 8; ++v) {
        float x = acc[v] * alpha + bv;
        if (ACT == 1)      x = fmaxf(x, 0.f);
        else if (ACT == 2) x = 0.5f * x * (1.f + erff(x * 0.70710678f));
        long off = base + (long)v * ldc;
        if (EPI == 0)      Cp[off] = x;
        else if (EPI == 1) Cp[off] = x + extra[v];
        else               Cp[off] = extra[v] + rs[v] * x;
    }
}

// ---- time embedding: te[b,:] = silu(t*Wt1 + bt1) @ Wt2 + bt2 ----
__global__ __launch_bounds__(256) void time_embed_kernel(
    const int* __restrict__ ts, const float* __restrict__ Wt1, const float* __restrict__ bt1,
    const float* __restrict__ Wt2, const float* __restrict__ bt2, float* __restrict__ te)
{
    __shared__ float s[Dm];
    int b = blockIdx.x;
    float t = (float)ts[b];
    for (int j = threadIdx.x; j < Dm; j += 256) {
        float x = t * Wt1[j] + bt1[j];
        s[j] = x / (1.f + expf(-x));
    }
    __syncthreads();
    for (int d = threadIdx.x; d < Dm; d += 256) {
        float acc = bt2[d];
        for (int j = 0; j < Dm; ++j) acc += s[j] * Wt2[j * Dm + d];
        te[(long)b * Dm + d] = acc;
    }
}

// ---- h = noisy_future @ W_in + b_in + te[b] ----
__global__ void input_proj_kernel(
    const float* __restrict__ nf, const float* __restrict__ W_in,
    const float* __restrict__ b_in, const float* __restrict__ te, float* __restrict__ h)
{
    long idx = (long)blockIdx.x * blockDim.x + threadIdx.x;
    if (idx >= (long)NTOK * Dm) return;
    int d = (int)(idx % Dm);
    long bs = idx / Dm;
    int b = (int)(bs / Sn);
    float acc = b_in[d] + te[(long)b * Dm + d];
    #pragma unroll
    for (int p = 0; p < PCn; ++p) acc += nf[bs * PCn + p] * W_in[p * Dm + d];
    h[idx] = acc;
}

// ---- row softmax over 512 elements, in place ----
__global__ __launch_bounds__(256) void softmax_rows_kernel(float* __restrict__ sc)
{
    __shared__ float red[256];
    long base = (long)blockIdx.x * Sn;
    int t = threadIdx.x;
    float a0 = sc[base + t], a1 = sc[base + t + 256];
    red[t] = fmaxf(a0, a1);
    __syncthreads();
    for (int s = 128; s > 0; s >>= 1) { if (t < s) red[t] = fmaxf(red[t], red[t + s]); __syncthreads(); }
    float m = red[0];
    __syncthreads();
    float e0 = expf(a0 - m), e1 = expf(a1 - m);
    red[t] = e0 + e1;
    __syncthreads();
    for (int s = 128; s > 0; s >>= 1) { if (t < s) red[t] += red[t + s]; __syncthreads(); }
    float inv = 1.f / red[0];
    sc[base + t] = e0 * inv;
    sc[base + t + 256] = e1 * inv;
}

// ---- layernorm over D=512, y = (x-mu)*rstd*g + b ----
__global__ __launch_bounds__(256) void layernorm_kernel(
    const float* __restrict__ x, const float* __restrict__ g,
    const float* __restrict__ bt, float* __restrict__ y)
{
    __shared__ float red[256];
    long base = (long)blockIdx.x * Dm;
    int t = threadIdx.x;
    float x0 = x[base + t], x1 = x[base + t + 256];
    red[t] = x0 + x1;
    __syncthreads();
    for (int s = 128; s > 0; s >>= 1) { if (t < s) red[t] += red[t + s]; __syncthreads(); }
    float mu = red[0] * (1.f / Dm);
    __syncthreads();
    float d0 = x0 - mu, d1 = x1 - mu;
    red[t] = d0 * d0 + d1 * d1;
    __syncthreads();
    for (int s = 128; s > 0; s >>= 1) { if (t < s) red[t] += red[t + s]; __syncthreads(); }
    float rstd = rsqrtf(red[0] * (1.f / Dm) + 1e-5f);
    y[base + t]       = d0 * rstd * g[t] + bt[t];
    y[base + t + 256] = d1 * rstd * g[t + 256] + bt[t + 256];
}

// ---- MoE gate: softmax(h @ Wg + bg), keep top-2 weights, zero others ----
__global__ void moe_gate_kernel(
    const float* __restrict__ h, const float* __restrict__ Wg,
    const float* __restrict__ bg, float* __restrict__ comb)
{
    int n = blockIdx.x * blockDim.x + threadIdx.x;
    if (n >= NTOK) return;
    float lg[En];
    #pragma unroll
    for (int e = 0; e < En; ++e) lg[e] = bg[e];
    const float* hp = h + (long)n * Dm;
    for (int d = 0; d < Dm; ++d) {
        float v = hp[d];
        #pragma unroll
        for (int e = 0; e < En; ++e) lg[e] += v * Wg[d * En + e];
    }
    float mx = lg[0];
    #pragma unroll
    for (int e = 1; e < En; ++e) mx = fmaxf(mx, lg[e]);
    float w[En], sum = 0.f;
    #pragma unroll
    for (int e = 0; e < En; ++e) { w[e] = expf(lg[e] - mx); sum += w[e]; }
    float inv = 1.f / sum;
    #pragma unroll
    for (int e = 0; e < En; ++e) w[e] *= inv;
    int i1 = 0;
    #pragma unroll
    for (int e = 1; e < En; ++e) if (w[e] > w[i1]) i1 = e;
    int i2 = (i1 == 0) ? 1 : 0;
    #pragma unroll
    for (int e = 0; e < En; ++e) if (e != i1 && w[e] > w[i2]) i2 = e;
    #pragma unroll
    for (int e = 0; e < En; ++e)
        comb[(long)n * En + e] = (e == i1) ? w[i1] : (e == i2 ? w[i2] : 0.f);
}

// ---- mean over context length ----
__global__ void ctx_mean_kernel(const float* __restrict__ ctx, float* __restrict__ cm)
{
    int idx = blockIdx.x * blockDim.x + threadIdx.x;
    if (idx >= Bn * Dm) return;
    int b = idx / Dm, d = idx % Dm;
    float acc = 0.f;
    for (int c = 0; c < CTXn; ++c) acc += ctx[((long)b * CTXn + c) * Dm + d];
    cm[idx] = acc * (1.f / CTXn);
}

// ---- out = (h + cmean[b]) @ W_out + b_out ----
__global__ void out_proj_kernel(
    const float* __restrict__ h, const float* __restrict__ cm,
    const float* __restrict__ W_out, const float* __restrict__ b_out, float* __restrict__ out)
{
    int idx = blockIdx.x * blockDim.x + threadIdx.x;
    if (idx >= NTOK * PCn) return;
    int p = idx % PCn;
    long bs = idx / PCn;
    int b = (int)(bs / Sn);
    float acc = b_out[p];
    const float* hp = h + bs * Dm;
    const float* cp = cm + (long)b * Dm;
    for (int d = 0; d < Dm; ++d) acc += (hp[d] + cp[d]) * W_out[d * PCn + p];
    out[idx] = acc;
}

extern "C" void kernel_launch(void* const* d_in, const int* in_sizes, int n_in,
                              void* d_out, int out_size, void* d_ws, size_t ws_size,
                              hipStream_t stream)
{
    (void)in_sizes; (void)n_in; (void)out_size; (void)ws_size;

    const float* context = (const float*)d_in[0];
    const float* noisy   = (const float*)d_in[1];
    const int*   tsteps  = (const int*)d_in[2];
    const float* W_in    = (const float*)d_in[3];
    const float* b_in    = (const float*)d_in[4];
    const float* W_out   = (const float*)d_in[5];
    const float* b_out   = (const float*)d_in[6];
    const float* Wt1     = (const float*)d_in[7];
    const float* bt1     = (const float*)d_in[8];
    const float* Wt2     = (const float*)d_in[9];
    const float* bt2     = (const float*)d_in[10];
    const float* Wqkv    = (const float*)d_in[11];
    const float* bqkv    = (const float*)d_in[12];
    const float* Wo      = (const float*)d_in[13];
    const float* bo      = (const float*)d_in[14];
    const float* ln1_g   = (const float*)d_in[15];
    const float* ln1_b   = (const float*)d_in[16];
    const float* W1      = (const float*)d_in[17];
    const float* b1      = (const float*)d_in[18];
    const float* W2      = (const float*)d_in[19];
    const float* b2      = (const float*)d_in[20];
    const float* ln2_g   = (const float*)d_in[21];
    const float* ln2_b   = (const float*)d_in[22];
    const float* Wg      = (const float*)d_in[23];
    const float* bg      = (const float*)d_in[24];
    const float* We1     = (const float*)d_in[25];
    const float* be1     = (const float*)d_in[26];
    const float* We2     = (const float*)d_in[27];
    const float* be2     = (const float*)d_in[28];
    float* out = (float*)d_out;

    // ---- workspace carve (floats). "big" is time-shared: scores / ffn hidden / moe hidden
    float* ws = (float*)d_ws;
    size_t off = 0;
    float* h    = ws + off; off += (size_t)NTOK * Dm;          // 2M
    float* qkv  = ws + off; off += (size_t)NTOK * 3 * Dm;      // 6M
    float* big  = ws + off; off += (size_t)Bn * Hn * Sn * Sn;  // 16.8M
    float* o    = ws + off; off += (size_t)NTOK * Dm;          // 2M
    float* u    = ws + off; off += (size_t)NTOK * Dm;          // 2M
    float* te   = ws + off; off += (size_t)Bn * Dm;
    float* comb = ws + off; off += (size_t)NTOK * En;
    float* cm   = ws + off; off += (size_t)Bn * Dm;

    const long sQ  = (long)Sn * 3 * Dm;   // per-batch stride inside qkv
    const long sSc = (long)Sn * Sn;       // per-head stride inside scores

    time_embed_kernel<<<Bn, 256, 0, stream>>>(tsteps, Wt1, bt1, Wt2, bt2, te);
    input_proj_kernel<<<((long)NTOK * Dm + 255) / 256, 256, 0, stream>>>(noisy, W_in, b_in, te, h);

    for (int l = 0; l < Ln; ++l) {
        const float* Wqkv_l = Wqkv + (size_t)l * Dm * 3 * Dm;
        const float* bqkv_l = bqkv + (size_t)l * 3 * Dm;
        const float* Wo_l   = Wo   + (size_t)l * Dm * Dm;
        const float* bo_l   = bo   + (size_t)l * Dm;
        const float* W1_l   = W1   + (size_t)l * Dm * DFn;
        const float* b1_l   = b1   + (size_t)l * DFn;
        const float* W2_l   = W2   + (size_t)l * DFn * Dm;
        const float* b2_l   = b2   + (size_t)l * Dm;

        // 1) qkv = h @ Wqkv + bqkv                      [4096 x 1536]
        gemm_wmma<0, 0, false, true><<<dim3(3 * Dm / BN, NTOK / BM, 1), 256, 0, stream>>>(
            h, Dm, Wqkv_l, 3 * Dm, bqkv_l, nullptr, 0, nullptr, 0,
            qkv, 3 * Dm, 3 * Dm, Dm, 1.f, 1, 0, 0, 0, 0, 0, 0);

        // 2) scores = Q @ K^T / 8 per (b,h)             [512 x 512, K=64] x64
        gemm_wmma<0, 0, true, false><<<dim3(Sn / BN, Sn / BM, Bn * Hn), 256, 0, stream>>>(
            qkv, 3 * Dm, qkv + Dm, 3 * Dm, nullptr, nullptr, 0, nullptr, 0,
            big, Sn, Sn, HDn, 0.125f,
            Hn, sQ, HDn, sQ, HDn, (long)Hn * sSc, sSc);

        // 3) softmax rows (in place)
        softmax_rows_kernel<<<Bn * Hn * Sn, 256, 0, stream>>>(big);

        // 4) o = P @ V per (b,h)                        [512 x 64, K=512] x64
        gemm_wmma<0, 0, false, false><<<dim3(HDn / BN, Sn / BM, Bn * Hn), 256, 0, stream>>>(
            big, Sn, qkv + 2 * Dm, 3 * Dm, nullptr, nullptr, 0, nullptr, 0,
            o, Dm, HDn, Sn, 1.f,
            Hn, (long)Hn * sSc, sSc, sQ, HDn, (long)Sn * Dm, HDn);

        // 5) u = o @ Wo + bo + h  (residual)            [4096 x 512]
        gemm_wmma<0, 1, false, true><<<dim3(Dm / BN, NTOK / BM, 1), 256, 0, stream>>>(
            o, Dm, Wo_l, Dm, bo_l, h, Dm, nullptr, 0,
            u, Dm, Dm, Dm, 1.f, 1, 0, 0, 0, 0, 0, 0);

        // 6) h = LN1(u)
        layernorm_kernel<<<NTOK, 256, 0, stream>>>(u, ln1_g + (size_t)l * Dm, ln1_b + (size_t)l * Dm, h);

        // 7) big = relu(h @ W1 + b1)                    [4096 x 2048]
        gemm_wmma<1, 0, false, true><<<dim3(DFn / BN, NTOK / BM, 1), 256, 0, stream>>>(
            h, Dm, W1_l, DFn, b1_l, nullptr, 0, nullptr, 0,
            big, DFn, DFn, Dm, 1.f, 1, 0, 0, 0, 0, 0, 0);

        // 8) u = big @ W2 + b2 + h
        gemm_wmma<0, 1, false, true><<<dim3(Dm / BN, NTOK / BM, 1), 256, 0, stream>>>(
            big, DFn, W2_l, Dm, b2_l, h, Dm, nullptr, 0,
            u, Dm, Dm, DFn, 1.f, 1, 0, 0, 0, 0, 0, 0);

        // 9) h = LN2(u)
        layernorm_kernel<<<NTOK, 256, 0, stream>>>(u, ln2_g + (size_t)l * Dm, ln2_b + (size_t)l * Dm, h);

        // 10) gate -> comb[4096 x 4]
        moe_gate_kernel<<<NTOK / 256, 256, 0, stream>>>(h, Wg, bg, comb);

        // 11) shared MoE: h += comb[:,e] * (gelu(h @ We1[e] + be1[e]) @ We2[e] + be2[e])
        for (int e = 0; e < En; ++e) {
            const float* We1_e = We1 + (size_t)e * Dm * DFn;
            const float* be1_e = be1 + (size_t)e * DFn;
            const float* We2_e = We2 + (size_t)e * DFn * Dm;
            const float* be2_e = be2 + (size_t)e * Dm;
            gemm_wmma<2, 0, false, true><<<dim3(DFn / BN, NTOK / BM, 1), 256, 0, stream>>>(
                h, Dm, We1_e, DFn, be1_e, nullptr, 0, nullptr, 0,
                big, DFn, DFn, Dm, 1.f, 1, 0, 0, 0, 0, 0, 0);
            gemm_wmma<0, 2, false, true><<<dim3(Dm / BN, NTOK / BM, 1), 256, 0, stream>>>(
                big, DFn, We2_e, Dm, be2_e, nullptr, 0, comb + e, En,
                h, Dm, Dm, DFn, 1.f, 1, 0, 0, 0, 0, 0, 0);
        }
    }

    ctx_mean_kernel<<<(Bn * Dm + 255) / 256, 256, 0, stream>>>(context, cm);
    out_proj_kernel<<<(NTOK * PCn + 255) / 256, 256, 0, stream>>>(h, cm, W_out, b_out, out);
}
